// CaptionModel_52089363366173
// MI455X (gfx1250) — matrix operand order repaired and checked by reading, not compiled, not run on hardware
//
#include <hip/hip_runtime.h>
#include <hip/hip_bf16.h>
#include <math.h>
#include <stdint.h>

// ---------------- problem constants ----------------
#define B_   32
#define H_   512
#define F_   2048
#define T_   200
#define TP_  208          // T padded to multiple of 16
#define MT_  (TP_ / 16)   // 13 t-tiles per batch row
#define V_   30522
#define VP_  30528        // V padded to multiple of 16
#define SOS_ 101
#define G3H_ (3 * H_)
#define NWG_GRU 8

typedef __attribute__((ext_vector_type(16))) __bf16 v16bf;
typedef __attribute__((ext_vector_type(8)))  __bf16 v8bf;
typedef __attribute__((ext_vector_type(8)))  float  v8f;
typedef __attribute__((ext_vector_type(4)))  unsigned int u32x4;
typedef __attribute__((ext_vector_type(8)))  int  i32x8;
typedef __attribute__((ext_vector_type(4)))  int  i32x4;

// ---------------- tiny prep kernels ----------------

// W_out (f32 [V,H]) -> bf16 [VP,H], zero pad rows
__global__ void k_cvt_wout(const float* __restrict__ W, __bf16* __restrict__ Wb) {
    size_t i = (size_t)blockIdx.x * blockDim.x + threadIdx.x;
    if (i >= (size_t)VP_ * H_) return;
    size_t v = i / H_;
    Wb[i] = (v < V_) ? (__bf16)W[i] : (__bf16)0.0f;
}

// W_ih (f32 [3H,H]) -> bf16
__global__ void k_cvt_wih(const float* __restrict__ W, __bf16* __restrict__ Wb) {
    int i = blockIdx.x * blockDim.x + threadIdx.x;
    if (i >= G3H_ * H_) return;
    Wb[i] = (__bf16)W[i];
}

// x0 = embed[SOS] broadcast over batch, bf16, into ping buffer 0
__global__ void k_init_x(const float* __restrict__ embed, __bf16* __restrict__ xb) {
    int i = blockIdx.x * blockDim.x + threadIdx.x;
    if (i >= B_ * H_) return;
    int k = i & (H_ - 1);
    xb[i] = (__bf16)embed[(size_t)SOS_ * H_ + k];
}

// h0 = feat @ W_hp^T + b_hp   (f32, one thread per output)
__global__ void k_h0(const float* __restrict__ feat, const float* __restrict__ Whp,
                     const float* __restrict__ bhp, float* __restrict__ h0) {
    int i = blockIdx.x * blockDim.x + threadIdx.x;
    if (i >= B_ * H_) return;
    int b = i / H_, o = i & (H_ - 1);
    const float4* fa = (const float4*)(feat + (size_t)b * F_);
    const float4* wa = (const float4*)(Whp + (size_t)o * F_);
    float acc = 0.0f;
    for (int k = 0; k < F_ / 4; ++k) {
        float4 x = fa[k], w = wa[k];
        acc += x.x * w.x + x.y * w.y + x.z * w.z + x.w * w.w;
    }
    h0[i] = acc + bhp[o];
}

// gh = h0 @ W_hh^T + b_hh, stored as [gate][b][j] (constant over all 200 steps)
__global__ void k_gh(const float* __restrict__ h0, const float* __restrict__ Whh,
                     const float* __restrict__ bhh, float* __restrict__ gh) {
    int i = blockIdx.x * blockDim.x + threadIdx.x;
    if (i >= B_ * G3H_) return;
    int b = i / G3H_, r = i % G3H_;
    int g = r / H_, j = r & (H_ - 1);
    const float4* ha = (const float4*)(h0 + (size_t)b * H_);
    const float4* wa = (const float4*)(Whh + (size_t)r * H_);
    float acc = 0.0f;
    for (int k = 0; k < H_ / 4; ++k) {
        float4 x = ha[k], w = wa[k];
        acc += x.x * w.x + x.y * w.y + x.z * w.z + x.w * w.w;
    }
    gh[((size_t)g * B_ + b) * H_ + j] = acc + bhh[r];
}

// ---------------- GRU recurrence: persistent, 8 WGs, grid barrier per step --------
// wave task: (dimtile in [0,32), bhalf in {0,1}) -> 16x16 output tile per gate.
// A = x_t tile (16 batch x 32 k, bf16), B = W_ih rows (bf16), f32 accumulate via WMMA.
__global__ void __launch_bounds__(256) k_gru(
    const __bf16* __restrict__ Wihb, const float* __restrict__ h0,
    const float* __restrict__ gh,    const float* __restrict__ b_ih,
    __bf16* __restrict__ xb,         __bf16* __restrict__ Xall,
    unsigned* __restrict__ bar) {

    const int lane = threadIdx.x & 31;
    const int wglb = blockIdx.x * 8 + (threadIdx.x >> 5);   // 0..63
    const int m0   = (wglb & 1) * 16;                       // batch half
    const int j0   = (wglb >> 1) * 16;                      // hidden-dim tile
    const int n    = lane & 15;
    const int hi   = lane >> 4;                             // lane group
    const int j    = j0 + n;

    // Hoisted step-invariant tiles (C layout: VGPR i -> M = i + hi*8, N = lane%16)
    float h0t[8], grt[8], gzt[8], gnt[8];
#pragma unroll
    for (int i = 0; i < 8; ++i) {
        int b = m0 + i + hi * 8;
        h0t[i] = h0[(size_t)b * H_ + j];
        grt[i] = gh[((size_t)0 * B_ + b) * H_ + j];
        gzt[i] = gh[((size_t)1 * B_ + b) * H_ + j];
        gnt[i] = gh[((size_t)2 * B_ + b) * H_ + j];
    }
    const float bir = b_ih[j], biz = b_ih[H_ + j], bin = b_ih[2 * H_ + j];

    const __bf16* Br = Wihb + (size_t)(0 * H_ + j) * H_ + hi * 16;
    const __bf16* Bz = Wihb + (size_t)(1 * H_ + j) * H_ + hi * 16;
    const __bf16* Bn = Wihb + (size_t)(2 * H_ + j) * H_ + hi * 16;
    const int aoff = hi * 16;
    const int mrow = m0 + n;                                // A row for this lane

    for (int t = 0; t < T_; ++t) {
        const int cur = t & 1, nxt = cur ^ 1;
        const __bf16* xrow = xb + ((size_t)cur * B_ + mrow) * H_;
        v8f ar = {}, az = {}, an = {};
#pragma unroll 4
        for (int kk = 0; kk < H_ / 32; ++kk) {
            const int k0 = kk * 32;
            v8bf a0 = *(const v8bf*)(xrow + k0 + aoff);
            v8bf a1 = *(const v8bf*)(xrow + k0 + aoff + 8);
            v16bf A = __builtin_shufflevector(a0, a1, 0,1,2,3,4,5,6,7,8,9,10,11,12,13,14,15);
            v16bf Bfr = *(const v16bf*)(Br + k0);
            v16bf Bfz = *(const v16bf*)(Bz + k0);
            v16bf Bfn = *(const v16bf*)(Bn + k0);
            ar = __builtin_amdgcn_wmma_f32_16x16x32_bf16(false, A, false, Bfr, (short)0, ar, false, false);
            az = __builtin_amdgcn_wmma_f32_16x16x32_bf16(false, A, false, Bfz, (short)0, az, false, false);
            an = __builtin_amdgcn_wmma_f32_16x16x32_bf16(false, A, false, Bfn, (short)0, an, false, false);
        }
        // gates + state update (f32), write bf16 copies for next step + output GEMM
#pragma unroll
        for (int i = 0; i < 8; ++i) {
            int b = m0 + i + hi * 8;
            float r  = 1.0f / (1.0f + __expf(-(ar[i] + bir + grt[i])));
            float z  = 1.0f / (1.0f + __expf(-(az[i] + biz + gzt[i])));
            float nn = tanhf(an[i] + bin + r * gnt[i]);
            float h  = (1.0f - z) * nn + z * h0t[i];
            Xall[((size_t)b * TP_ + t) * H_ + j]      = (__bf16)h;
            xb[((size_t)nxt * B_ + b) * H_ + j]       = (__bf16)h;
        }
        // grid-wide barrier (8 co-resident WGs), monotonic counter
        __syncthreads();
        if (threadIdx.x == 0) {
            __threadfence();
            atomicAdd(bar, 1u);
            const unsigned tgt = (unsigned)NWG_GRU * (unsigned)(t + 1);
            while (__hip_atomic_load(bar, __ATOMIC_RELAXED, __HIP_MEMORY_SCOPE_AGENT) < tgt)
                __builtin_amdgcn_s_sleep(2);
            __threadfence();
        }
        __syncthreads();
    }
}

// ---------------- output projection: logits[b][v][t] = X @ W_out^T + b_out -------
// One WG per (batch b, 8 vocab tiles). The whole 208x512 bf16 A-panel of batch b
// (208 KB, contiguous) is DMA'd into LDS with the Tensor Data Mover; each wave
// then reuses every streamed B fragment across all 13 t-tiles (13 WMMAs per
// 32B B-load -> ~13x less L2 traffic on the 31 MB W_out panel).
__global__ void __launch_bounds__(256) k_logits(
    const __bf16* __restrict__ Xall, const __bf16* __restrict__ Woutb,
    const float* __restrict__ bout,  float* __restrict__ out) {

    extern __shared__ __bf16 As[];                          // TP_*H_ bf16 = 208 KB
    const int b = blockIdx.y;

    if (threadIdx.x < 32) {                                 // wave 0 issues the TDM
        const __bf16* gp = Xall + (size_t)b * TP_ * H_;
        const unsigned long long ga = (unsigned long long)(uintptr_t)gp;
        const unsigned lds = (unsigned)(uintptr_t)(void*)As;
        // D# group0: count=1 | lds_addr | global_addr[56:0] | type=2
        u32x4 g0 = { 1u, lds,
                     (unsigned)(ga & 0xFFFFFFFFu),
                     (unsigned)((ga >> 32) & 0x01FFFFFFu) | 0x80000000u };
        // D# group1: data_size=4B; 2D tensor/tile 4096 x 13 dwords, stride 4096
        i32x8 g1 = { (int)(2u << 16),        // workgroup_mask=0, data_size=2 (4B)
                     (int)0x10000000,        // tensor_dim0 = 4096 (low16 in [31:16])
                     (int)(13 << 16),        // tensor_dim0 hi=0, tensor_dim1 = 13
                     (int)0x10000000,        // tensor_dim1 hi=0, tile_dim0 = 4096
                     13,                     // tile_dim1 = 13, tile_dim2 = 0
                     4096,                   // tensor_dim0_stride lo
                     (int)0x10000000,        // dim0_stride hi=0, dim1_stride lo=4096
                     0 };                    // dim1_stride hi = 0
        i32x4 gz  = { 0, 0, 0, 0 };          // D# groups 2/3 unused (<=2D tensor)
        i32x8 gz8 = { 0, 0, 0, 0, 0, 0, 0, 0 };
        __builtin_amdgcn_tensor_load_to_lds(g0, g1, gz, gz, gz8, 0);
        __builtin_amdgcn_s_wait_tensorcnt(0);               // TENSORcnt -> 0
    }
    __syncthreads();

    const int lane = threadIdx.x & 31;
    const int nt   = blockIdx.x * 8 + (threadIdx.x >> 5);
    if (nt >= VP_ / 16) return;
    const int n0 = nt * 16;
    const int n  = lane & 15;
    const int hi = lane >> 4;
    const int v  = n0 + n;

    const __bf16* arow = As + (size_t)n * H_;               // row M = lane%16 within a tile
    const __bf16* brow = Woutb + (size_t)v * H_ + hi * 16;  // B col = lane%16, contiguous K
    const int aoff = hi * 16;

    __builtin_prefetch(brow, 0, 1);                         // global_prefetch_b8 on W_out row

    v8f acc[MT_] = {};
    for (int kk = 0; kk < H_ / 32; ++kk) {
        const int k0 = kk * 32;
        v16bf Bf = *(const v16bf*)(brow + k0);              // one 32B B frag ...
#pragma unroll
        for (int m = 0; m < MT_; ++m) {                     // ... feeds 13 WMMAs
            const __bf16* ap = arow + (size_t)m * 16 * H_ + k0 + aoff;
            v8bf a0 = *(const v8bf*)(ap);
            v8bf a1 = *(const v8bf*)(ap + 8);
            v16bf A = __builtin_shufflevector(a0, a1, 0,1,2,3,4,5,6,7,8,9,10,11,12,13,14,15);
            acc[m] = __builtin_amdgcn_wmma_f32_16x16x32_bf16(false, A, false, Bf, (short)0, acc[m], false, false);
        }
    }

    if (v < V_) {
        const float bias = bout[v];
        float* obase = out + ((size_t)b * V_ + v) * T_;
#pragma unroll
        for (int m = 0; m < MT_; ++m) {
            const int tbase = m * 16 + hi * 8;              // 8 consecutive t per lane
            if (tbase + 8 <= T_) {                          // contiguous, 32B-aligned
                float4 p0 = make_float4(acc[m][0]+bias, acc[m][1]+bias, acc[m][2]+bias, acc[m][3]+bias);
                float4 p1 = make_float4(acc[m][4]+bias, acc[m][5]+bias, acc[m][6]+bias, acc[m][7]+bias);
                *(float4*)(obase + tbase)     = p0;
                *(float4*)(obase + tbase + 4) = p1;
            } else {
#pragma unroll
                for (int i = 0; i < 8; ++i)
                    if (tbase + i < T_) obase[tbase + i] = acc[m][i] + bias;
            }
        }
    }
}

// ---------------- host launch ----------------
extern "C" void kernel_launch(void* const* d_in, const int* in_sizes, int n_in,
                              void* d_out, int out_size, void* d_ws, size_t ws_size,
                              hipStream_t stream) {
    const float* feat  = (const float*)d_in[0];
    const float* W_hp  = (const float*)d_in[1];
    const float* b_hp  = (const float*)d_in[2];
    const float* W_ih  = (const float*)d_in[3];
    const float* W_hh  = (const float*)d_in[4];
    const float* b_ih  = (const float*)d_in[5];
    const float* b_hh  = (const float*)d_in[6];
    const float* embed = (const float*)d_in[7];
    const float* W_out = (const float*)d_in[8];
    const float* b_out = (const float*)d_in[9];
    float* out = (float*)d_out;

    // workspace carve-out (~40 MB total), 256B aligned slices
    char* ws = (char*)d_ws;
    size_t off = 0;
    auto carve = [&](size_t bytes) -> void* {
        void* p = ws + off;
        off = (off + bytes + 255) & ~(size_t)255;
        return p;
    };
    unsigned* bar   = (unsigned*)carve(256);
    __bf16*   xb    = (__bf16*)carve((size_t)2 * B_ * H_ * 2);       // ping-pong state
    float*    h0    = (float*)carve((size_t)B_ * H_ * 4);
    float*    gh    = (float*)carve((size_t)B_ * G3H_ * 4);
    __bf16*   Wihb  = (__bf16*)carve((size_t)G3H_ * H_ * 2);
    __bf16*   Woutb = (__bf16*)carve((size_t)VP_ * H_ * 2);
    __bf16*   Xall  = (__bf16*)carve((size_t)B_ * TP_ * H_ * 2);
    (void)ws_size; (void)in_sizes; (void)n_in; (void)out_size;

    (void)hipMemsetAsync(bar, 0, 256, stream);                        // barrier reset (graph-safe)

    k_cvt_wout<<<((size_t)VP_ * H_ + 255) / 256, 256, 0, stream>>>(W_out, Woutb);
    k_cvt_wih <<<(G3H_ * H_ + 255) / 256,        256, 0, stream>>>(W_ih, Wihb);
    k_init_x  <<<(B_ * H_ + 255) / 256,          256, 0, stream>>>(embed, xb);
    k_h0      <<<(B_ * H_ + 255) / 256,          256, 0, stream>>>(feat, W_hp, b_hp, h0);
    k_gh      <<<(B_ * G3H_ + 255) / 256,        256, 0, stream>>>(h0, W_hh, b_hh, gh);

    k_gru<<<NWG_GRU, 256, 0, stream>>>(Wihb, h0, gh, b_ih, xb, Xall, bar);

    dim3 grid((VP_ / 16 + 7) / 8, B_);                                // 239 x 32
    k_logits<<<grid, 256, (size_t)TP_ * H_ * 2, stream>>>(Xall, Woutb, b_out, out);
}
